// PNA_8598524527292
// MI455X (gfx1250) — matrix operand order repaired and checked by reading, not compiled, not run on hardware
//
#include <hip/hip_runtime.h>
#include <hip/hip_bf16.h>

#define N_NODES 50000
#define N_EDGES 600000
#define N_GRAPHS 64
#define D_HID 128
#define D_LAT 64
#define D_CAT (13 * 128)   // 1664

typedef __bf16 v16bf __attribute__((ext_vector_type(16)));
typedef __bf16 v8bf  __attribute__((ext_vector_type(8)));
typedef float  v8f   __attribute__((ext_vector_type(8)));

// ---- order-preserving float <-> uint encoding for atomic max/min ----
__device__ __forceinline__ unsigned fenc(float f) {
    unsigned u = __float_as_uint(f);
    return (u & 0x80000000u) ? ~u : (u | 0x80000000u);
}
__device__ __forceinline__ float fdec(unsigned u) {
    u = (u & 0x80000000u) ? (u & 0x7FFFFFFFu) : ~u;
    return __uint_as_float(u);
}
#define ENC_NEG_INF 0x007FFFFFu   // fenc(-inf)
#define ENC_POS_INF 0xFF800000u   // fenc(+inf)

// -------------------- utility kernels --------------------
__global__ void k_zero_f32(float* p, long n) {
    long i = (long)blockIdx.x * blockDim.x + threadIdx.x;
    if (i < n) p[i] = 0.0f;
}

__global__ void k_reset_agg(float* sum, float* ssq, unsigned* mx, unsigned* mn, long n) {
    long i = (long)blockIdx.x * blockDim.x + threadIdx.x;
    if (i < n) { sum[i] = 0.0f; ssq[i] = 0.0f; mx[i] = ENC_NEG_INF; mn[i] = ENC_POS_INF; }
}

__global__ void k_deg(const int* __restrict__ dst, float* __restrict__ deg, int E) {
    int i = blockIdx.x * blockDim.x + threadIdx.x;
    if (i < E) atomicAdd(&deg[dst[i]], 1.0f);
}

__global__ void k_logsum(const float* __restrict__ deg, float* __restrict__ logsum, int n) {
    __shared__ float s[256];
    int tid = threadIdx.x;
    int i = blockIdx.x * 256 + tid;
    s[tid] = (i < n) ? logf(deg[i] + 1.0f) : 0.0f;
    __syncthreads();
    for (int st = 128; st > 0; st >>= 1) {
        if (tid < st) s[tid] += s[tid + st];
        __syncthreads();
    }
    if (tid == 0) atomicAdd(logsum, s[0]);
}

__global__ void k_ampatt(const float* __restrict__ deg, const float* __restrict__ logsum,
                         float* __restrict__ amp, float* __restrict__ att, int n) {
    int i = blockIdx.x * blockDim.x + threadIdx.x;
    if (i >= n) return;
    float delta = logsum[0] / (float)N_NODES;
    float ld = logf(deg[i] + 1.0f);
    amp[i] = ld / delta;
    att[i] = delta / fmaxf(ld, 1e-5f);
}

// Convert all three W matrices to column-major bf16: Wt[l][c*D_CAT + k] = W_l[k*128 + c]
__global__ void k_convW(const float* __restrict__ W0, const float* __restrict__ W1,
                        const float* __restrict__ W2, __bf16* __restrict__ Wt) {
    long gid = (long)blockIdx.x * blockDim.x + threadIdx.x;
    long total = 3L * D_CAT * D_HID;
    if (gid >= total) return;
    int l = (int)(gid / (D_CAT * D_HID));
    long rem = gid - (long)l * (D_CAT * D_HID);
    int k = (int)(rem / D_HID);
    int c = (int)(rem - (long)k * D_HID);
    const float* W = (l == 0) ? W0 : (l == 1) ? W1 : W2;
    Wt[(size_t)l * D_HID * D_CAT + (size_t)c * D_CAT + k] = (__bf16)W[(size_t)k * D_HID + c];
}

// -------------------- edge scatter: sum / sumsq / max / min --------------------
__global__ __launch_bounds__(256) void k_scatter(
        const float* __restrict__ h, const int* __restrict__ src, const int* __restrict__ dst,
        float* __restrict__ sum, float* __restrict__ ssq,
        unsigned* __restrict__ mx, unsigned* __restrict__ mn, int E) {
    long gid = (long)blockIdx.x * blockDim.x + threadIdx.x;
    if (gid >= (long)E * 32) return;
    int e = (int)(gid >> 5);
    int c = (int)(gid & 31);
    int s = src[e], d = dst[e];
    const float4 m = *(const float4*)(h + (size_t)s * 128 + c * 4);
    size_t o = (size_t)d * 128 + c * 4;
    atomicAdd(sum + o + 0, m.x);
    atomicAdd(sum + o + 1, m.y);
    atomicAdd(sum + o + 2, m.z);
    atomicAdd(sum + o + 3, m.w);
    atomicAdd(ssq + o + 0, m.x * m.x);
    atomicAdd(ssq + o + 1, m.y * m.y);
    atomicAdd(ssq + o + 2, m.z * m.z);
    atomicAdd(ssq + o + 3, m.w * m.w);
    atomicMax(mx + o + 0, fenc(m.x));
    atomicMax(mx + o + 1, fenc(m.y));
    atomicMax(mx + o + 2, fenc(m.z));
    atomicMax(mx + o + 3, fenc(m.w));
    atomicMin(mn + o + 0, fenc(m.x));
    atomicMin(mn + o + 1, fenc(m.y));
    atomicMin(mn + o + 2, fenc(m.z));
    atomicMin(mn + o + 3, fenc(m.w));
}

// -------------------- build bf16 base blocks [N,5,128]: h, mean, max, min, std -----
__global__ void k_build_base(const float* __restrict__ h, const float* __restrict__ deg,
                             const float* __restrict__ sum, const float* __restrict__ ssq,
                             const unsigned* __restrict__ mx, const unsigned* __restrict__ mn,
                             __bf16* __restrict__ base) {
    long gid = (long)blockIdx.x * blockDim.x + threadIdx.x;
    if (gid >= (long)N_NODES * 128) return;
    int n = (int)(gid >> 7);
    int d = (int)(gid & 127);
    float dg = deg[n];
    float cnt = fmaxf(dg, 1.0f);
    float mean = sum[gid] / cnt;
    float var = fmaxf(ssq[gid] / cnt - mean * mean, 0.0f);
    float stdv = sqrtf(var + 1e-5f);
    bool has = dg > 0.0f;
    float mxv = has ? fdec(mx[gid]) : 0.0f;
    float mnv = has ? fdec(mn[gid]) : 0.0f;
    __bf16* b = base + (size_t)n * 640;
    b[d]       = (__bf16)h[gid];
    b[128 + d] = (__bf16)mean;
    b[256 + d] = (__bf16)mxv;
    b[384 + d] = (__bf16)mnv;
    b[512 + d] = (__bf16)stdv;
}

// -------------------- WMMA GEMM: out[N,128] = relu(feat[N,1664] @ W + b) --------
// feat = [h, agg, agg*amp, agg*att]. Since amp/att are per-row scalars:
//   feat @ W = [h,agg] @ W[0:640] + amp .* (agg @ W[640:1152]) + att .* (agg @ W[1152:1664])
// so A fragments are pure bf16 loads and scaling happens once in the f32 epilogue.
// One workgroup (8 waves) per 16-row strip (50000 = 16*3125); wave w -> cols [16w,16w+16).
__global__ __launch_bounds__(256) void k_gemm(
        const __bf16* __restrict__ base, const float* __restrict__ amp,
        const float* __restrict__ att, const __bf16* __restrict__ Wt,
        const float* __restrict__ bias, float* __restrict__ out) {
    const int lane = threadIdx.x & 31;
    const int wave = threadIdx.x >> 5;          // 0..7 -> column tile
    const int m0 = blockIdx.x * 16;             // row strip
    const int half = lane >> 4;                 // 0 or 1
    const int l16 = lane & 15;

    // A fragment: lanes 0-15 -> K local {0..15},
    // lanes 16-31 -> K local {16..31}: two contiguous 16B chunks per lane.
    const __bf16* abase = base + (size_t)(m0 + l16) * 640 + half * 16;
    // B layout: lanes 0-15 hold K=k0..k0+15 (col=lane), lanes 16-31 K=k0+16..k0+31.
    const __bf16* wcol = Wt + (size_t)(wave * 16 + l16) * D_CAT + half * 16;

    auto loadA = [&](int off) -> v16bf {
        v8bf c0 = *(const v8bf*)(abase + off);
        v8bf c1 = *(const v8bf*)(abase + off + 8);
        return __builtin_shufflevector(c0, c1, 0, 1, 2, 3, 4, 5, 6, 7,
                                               8, 9, 10, 11, 12, 13, 14, 15);
    };
    auto loadB = [&](int k) -> v16bf { return *(const v16bf*)(wcol + k); };

    v8f acc0 = {}, acc1 = {}, acc2 = {};

    // block 0: h columns (K = 0..127)
#pragma unroll
    for (int s = 0; s < 4; ++s) {
        v16bf a = loadA(32 * s);
        acc0 = __builtin_amdgcn_wmma_f32_16x16x32_bf16(
            false, a, false, loadB(32 * s), (short)0, acc0, false, false);
    }
    // blocks 1..4: agg columns -> feed 3 accumulators (raw / amp-scaled / att-scaled W cols)
#pragma unroll
    for (int b = 1; b <= 4; ++b) {
#pragma unroll
        for (int s = 0; s < 4; ++s) {
            const int dd = b * 128 + 32 * s;
            v16bf a = loadA(dd);
            acc0 = __builtin_amdgcn_wmma_f32_16x16x32_bf16(
                false, a, false, loadB(dd), (short)0, acc0, false, false);
            acc1 = __builtin_amdgcn_wmma_f32_16x16x32_bf16(
                false, a, false, loadB(dd + 512), (short)0, acc1, false, false);
            acc2 = __builtin_amdgcn_wmma_f32_16x16x32_bf16(
                false, a, false, loadB(dd + 1024), (short)0, acc2, false, false);
        }
    }

    // C layout: lanes 0-15 -> rows M=0..7 in acc[0..7] (col=l16); lanes 16-31 -> M=8..15.
    const int ccol = wave * 16 + l16;
    const float bv = bias[ccol];
#pragma unroll
    for (int r = 0; r < 8; ++r) {
        int m = m0 + half * 8 + r;
        float v = acc0[r] + amp[m] * acc1[r] + att[m] * acc2[r] + bv;
        out[(size_t)m * 128 + ccol] = v > 0.0f ? v : 0.0f;
    }
}

// -------------------- pooling / batchnorm / final FC --------------------
__global__ void k_pool(const float* __restrict__ h, const int* __restrict__ batch,
                       float* __restrict__ pooled) {
    long gid = (long)blockIdx.x * blockDim.x + threadIdx.x;
    if (gid >= (long)N_NODES * 128) return;
    int n = (int)(gid >> 7);
    int d = (int)(gid & 127);
    atomicAdd(pooled + (size_t)batch[n] * 128 + d, h[gid]);
}

__global__ void k_bnstats(const float* __restrict__ pooled, float* __restrict__ mu,
                          float* __restrict__ var) {
    int j = threadIdx.x;   // 128 threads
    float s = 0.0f, s2 = 0.0f;
    for (int g = 0; g < N_GRAPHS; ++g) {
        float v = pooled[(size_t)g * 128 + j];
        s += v; s2 += v * v;
    }
    float m = s / (float)N_GRAPHS;
    mu[j] = m;
    var[j] = s2 / (float)N_GRAPHS - m * m;
}

__global__ void k_final(const float* __restrict__ pooled, const float* __restrict__ mu,
                        const float* __restrict__ var, const float* __restrict__ gamma,
                        const float* __restrict__ beta, const float* __restrict__ fcW,
                        const float* __restrict__ fcb, float* __restrict__ out) {
    int g = blockIdx.x;     // 64
    int j = threadIdx.x;    // 64
    float acc = fcb[j];
    for (int k = 0; k < 128; ++k) {
        float bn = (pooled[(size_t)g * 128 + k] - mu[k]) * rsqrtf(var[k] + 1e-5f) * gamma[k] + beta[k];
        acc += bn * fcW[(size_t)k * D_LAT + j];
    }
    out[(size_t)g * D_LAT + j] = acc;
}

// -------------------- host launcher --------------------
extern "C" void kernel_launch(void* const* d_in, const int* in_sizes, int n_in,
                              void* d_out, int out_size, void* d_ws, size_t ws_size,
                              hipStream_t stream) {
    const float* x     = (const float*)d_in[0];
    const int*   ei    = (const int*)d_in[1];
    const int*   src   = ei;
    const int*   dst   = ei + N_EDGES;
    const int*   batch = (const int*)d_in[2];
    const float* W0 = (const float*)d_in[3];
    const float* b0 = (const float*)d_in[4];
    const float* W1 = (const float*)d_in[5];
    const float* b1 = (const float*)d_in[6];
    const float* W2 = (const float*)d_in[7];
    const float* b2 = (const float*)d_in[8];
    const float* gamma = (const float*)d_in[9];
    const float* beta  = (const float*)d_in[10];
    const float* fcW   = (const float*)d_in[11];
    const float* fcb   = (const float*)d_in[12];
    float* out = (float*)d_out;

    // ---- carve workspace ----
    char* p = (char*)d_ws;
    size_t off = 0;
    auto carve = [&](size_t bytes) -> void* {
        void* r = p + off;
        off = (off + bytes + 255) & ~(size_t)255;
        return r;
    };
    float*    deg    = (float*)carve((size_t)N_NODES * 4);
    float*    logsum = (float*)carve(256);
    float*    amp    = (float*)carve((size_t)N_NODES * 4);
    float*    att    = (float*)carve((size_t)N_NODES * 4);
    float*    sum    = (float*)carve((size_t)N_NODES * 128 * 4);
    float*    ssq    = (float*)carve((size_t)N_NODES * 128 * 4);
    unsigned* mx     = (unsigned*)carve((size_t)N_NODES * 128 * 4);
    unsigned* mn     = (unsigned*)carve((size_t)N_NODES * 128 * 4);
    __bf16*   base   = (__bf16*)carve((size_t)N_NODES * 640 * 2);
    float*    hA     = (float*)carve((size_t)N_NODES * 128 * 4);
    float*    hB     = (float*)carve((size_t)N_NODES * 128 * 4);
    __bf16*   Wt     = (__bf16*)carve((size_t)3 * D_HID * D_CAT * 2);
    float*    pooled = (float*)carve((size_t)N_GRAPHS * 128 * 4);
    float*    mu     = (float*)carve(512);
    float*    var    = (float*)carve(512);

    const int T = 256;
    auto cdiv = [](long a, long b) -> unsigned { return (unsigned)((a + b - 1) / b); };
    const long NC = (long)N_NODES * 128;

    // ---- graph-wide precompute ----
    k_zero_f32<<<cdiv(N_NODES, T), T, 0, stream>>>(deg, N_NODES);
    k_zero_f32<<<1, 64, 0, stream>>>(logsum, 1);
    k_zero_f32<<<cdiv(N_GRAPHS * 128, T), T, 0, stream>>>(pooled, N_GRAPHS * 128);
    k_deg<<<cdiv(N_EDGES, T), T, 0, stream>>>(dst, deg, N_EDGES);
    k_logsum<<<cdiv(N_NODES, T), T, 0, stream>>>(deg, logsum, N_NODES);
    k_ampatt<<<cdiv(N_NODES, T), T, 0, stream>>>(deg, logsum, amp, att, N_NODES);
    k_convW<<<cdiv(3L * D_CAT * D_HID, T), T, 0, stream>>>(W0, W1, W2, Wt);

    // ---- 3 PNA layers ----
    const float* biases[3] = {b0, b1, b2};
    const float* hcur = x;
    float* bufs[2] = {hA, hB};
    for (int l = 0; l < 3; ++l) {
        float* hout = bufs[l & 1];
        k_reset_agg<<<cdiv(NC, T), T, 0, stream>>>(sum, ssq, mx, mn, NC);
        k_scatter<<<cdiv((long)N_EDGES * 32, T), T, 0, stream>>>(hcur, src, dst, sum, ssq, mx, mn, N_EDGES);
        k_build_base<<<cdiv(NC, T), T, 0, stream>>>(hcur, deg, sum, ssq, mx, mn, base);
        k_gemm<<<N_NODES / 16, 256, 0, stream>>>(base, amp, att,
                                                 Wt + (size_t)l * D_HID * D_CAT,
                                                 biases[l], hout);
        hcur = hout;
    }

    // ---- pool + batchnorm + final FC ----
    k_pool<<<cdiv(NC, T), T, 0, stream>>>(hcur, batch, pooled);
    k_bnstats<<<1, 128, 0, stream>>>(pooled, mu, var);
    k_final<<<N_GRAPHS, D_LAT, 0, stream>>>(pooled, mu, var, gamma, beta, fcW, fcb, out);
}
